// BasicNonLocalBlock_64570538328627
// MI455X (gfx1250) — matrix operand order repaired and checked by reading, not compiled, not run on hardware
//
#include <hip/hip_runtime.h>
#include <hip/hip_bf16.h>

typedef __bf16 bf16;
typedef __attribute__((ext_vector_type(8)))  __bf16 v8bf;
typedef __attribute__((ext_vector_type(16))) __bf16 v16bf;
typedef __attribute__((ext_vector_type(8)))  float  v8f;

#define NB  4
#define CIN 512
#define KCH 256
#define VCH 256
#define OCH 512
#define LL  4096

#define WMMA_BF16(a, b, c) \
  __builtin_amdgcn_wmma_f32_16x16x32_bf16(false, (a), false, (b), (short)0, (c), false, false)

// ---- fragment loaders -------------------------------------------------------
// A matrix (M x K, 16-bit): per lane row m = lane&15; elements are
// K {kbase+0..7, kbase+16..23} for lanes 0-15 and {+8..15, +24..31} for 16-31.
__device__ __forceinline__ v16bf load_frag_a(const bf16* rowptr, int kbase, int half) {
  const bf16* p = rowptr + kbase + half * 8;
  v8bf lo = *(const v8bf*)(p);
  v8bf hi = *(const v8bf*)(p + 16);
  v16bf r;
#pragma unroll
  for (int i = 0; i < 8; ++i) { r[i] = lo[i]; r[i + 8] = hi[i]; }
  return r;
}

// B matrix (K x N, 16-bit): per lane column n = lane&15; elements are
// K {kbase+0..7, kbase+16..23} for lanes 0-15 and {+8..15, +24..31} for 16-31.
__device__ __forceinline__ v16bf load_frag_b(const bf16* rowptr, int kbase, int half) {
  const bf16* p = rowptr + kbase + half * 8;
  v8bf lo = *(const v8bf*)(p);
  v8bf hi = *(const v8bf*)(p + 16);
  v16bf r;
#pragma unroll
  for (int i = 0; i < 8; ++i) { r[i] = lo[i]; r[i + 8] = hi[i]; }
  return r;
}

// ---- gfx1250 async global->LDS copy (ASYNCcnt-tracked) ----------------------
__device__ __forceinline__ void async_copy_b128(uint32_t lds_addr, const bf16* gaddr) {
  asm volatile("global_load_async_to_lds_b128 %0, %1, off"
               :
               : "v"(lds_addr), "v"((unsigned long long)(uintptr_t)gaddr)
               : "memory");
}

// ---- prep: fold BN into weights, convert to bf16 ----------------------------
__global__ __launch_bounds__(256) void fold_weights(
    const float* __restrict__ wq, const float* __restrict__ bq,
    const float* __restrict__ gq, const float* __restrict__ betaq,
    const float* __restrict__ mq, const float* __restrict__ vq,
    const float* __restrict__ wk, const float* __restrict__ bk,
    const float* __restrict__ gk, const float* __restrict__ betak,
    const float* __restrict__ mk, const float* __restrict__ vk,
    const float* __restrict__ wv, const float* __restrict__ bv,
    const float* __restrict__ wW,
    bf16* __restrict__ Wqkv, float* __restrict__ biasAll, bf16* __restrict__ wWb) {
  int b = blockIdx.x;
  int t = threadIdx.x;
  if (b < 768) {
    int o = b;
    const float* wsrc;
    float scale, bias;
    if (o < 256) {
      float inv = gq[o] * rsqrtf(vq[o] + 1e-5f);
      wsrc = wq + (size_t)o * CIN;
      scale = inv;
      bias = bq[o] * inv + betaq[o] - mq[o] * inv;
    } else if (o < 512) {
      int oo = o - 256;
      float inv = gk[oo] * rsqrtf(vk[oo] + 1e-5f);
      wsrc = wk + (size_t)oo * CIN;
      scale = inv;
      bias = bk[oo] * inv + betak[oo] - mk[oo] * inv;
    } else {
      int oo = o - 512;
      wsrc = wv + (size_t)oo * CIN;
      scale = 1.0f;
      bias = bv[oo];
    }
    for (int c = t; c < CIN; c += 256) Wqkv[(size_t)o * CIN + c] = (bf16)(wsrc[c] * scale);
    if (t == 0) biasAll[o] = bias;
  } else {
    int o = b - 768;  // 0..511 rows of wW [OCH, VCH]
    for (int c = t; c < VCH; c += 256) wWb[(size_t)o * VCH + c] = (bf16)wW[(size_t)o * VCH + c];
  }
}

// ---- prep: x [N,C,L] fp32 -> Xt [N,L,C] bf16 --------------------------------
__global__ __launch_bounds__(256) void transpose_x(const float* __restrict__ x,
                                                   bf16* __restrict__ Xt) {
  int nc = blockIdx.x;           // N*CIN blocks
  int n = nc >> 9, c = nc & 511;
  const float* xr = x + ((size_t)n * CIN + c) * LL;
  bf16* xt = Xt + (size_t)n * LL * CIN;
  for (int l = threadIdx.x; l < LL; l += blockDim.x)
    xt[(size_t)l * CIN + c] = (bf16)xr[l];
}

// ---- QKV projection: Out[768,4096] = Wqkv[768,512] * X[512,4096] ------------
__global__ __launch_bounds__(256) void proj_kernel(
    const bf16* __restrict__ Xt, const bf16* __restrict__ Wqkv,
    const float* __restrict__ biasAll,
    bf16* __restrict__ Qt, bf16* __restrict__ Kt, bf16* __restrict__ Vm) {
  int lt = blockIdx.x;   // 64 tiles of 64 columns (l)
  int ot = blockIdx.y;   // 6 tiles of 128 rows (o)
  int n  = blockIdx.z;
  int wave = threadIdx.x >> 5;
  int lane = threadIdx.x & 31;
  int half = lane >> 4;
  int l15 = lane & 15;
  int obase = ot * 128 + wave * 16;
  int lbase = lt * 64;

  const bf16* Xn = Xt + (size_t)n * LL * CIN;
  const bf16* arow = Wqkv + (size_t)(obase + l15) * CIN;
  const bf16* brow[4];
#pragma unroll
  for (int t = 0; t < 4; ++t) brow[t] = Xn + (size_t)(lbase + t * 16 + l15) * CIN;

  v8f acc[4] = {};
  for (int kc = 0; kc < CIN; kc += 32) {
    v16bf a = load_frag_a(arow, kc, half);
#pragma unroll
    for (int t = 0; t < 4; ++t) {
      v16bf b = load_frag_b(brow[t], kc, half);
      acc[t] = WMMA_BF16(a, b, acc[t]);
    }
  }

  float bias[8];
#pragma unroll
  for (int r = 0; r < 8; ++r) bias[r] = biasAll[obase + half * 8 + r];

  if (obase < 256) {
    bf16* dst = Qt + (size_t)n * LL * KCH;
#pragma unroll
    for (int t = 0; t < 4; ++t)
#pragma unroll
      for (int r = 0; r < 8; ++r) {
        int o = obase + half * 8 + r;
        int l = lbase + t * 16 + l15;
        dst[(size_t)l * KCH + o] = (bf16)(acc[t][r] + bias[r]);
      }
  } else if (obase < 512) {
    bf16* dst = Kt + (size_t)n * LL * KCH;
#pragma unroll
    for (int t = 0; t < 4; ++t)
#pragma unroll
      for (int r = 0; r < 8; ++r) {
        int o = obase - 256 + half * 8 + r;
        int l = lbase + t * 16 + l15;
        dst[(size_t)l * KCH + o] = (bf16)(acc[t][r] + bias[r]);
      }
  } else {
    bf16* dst = Vm + (size_t)n * VCH * LL;
#pragma unroll
    for (int t = 0; t < 4; ++t)
#pragma unroll
      for (int r = 0; r < 8; ++r) {
        int o = obase - 512 + half * 8 + r;
        int l = lbase + t * 16 + l15;
        dst[(size_t)o * LL + l] = (bf16)(acc[t][r] + bias[r]);
      }
  }
}

// ---- fused flash attention with async double-buffered K/V LDS staging -------
__global__ __launch_bounds__(128) void attn_kernel(
    const bf16* __restrict__ Qt, const bf16* __restrict__ Kt,
    const bf16* __restrict__ Vm, bf16* __restrict__ Ctx) {
  int qt = blockIdx.x;   // 64 query tiles of 64
  int n  = blockIdx.y;
  int tid  = threadIdx.x;
  int wave = tid >> 5;
  int lane = tid & 31;
  int half = lane >> 4;
  int l15 = lane & 15;
  int qbase = qt * 64 + wave * 16;

  // double-buffered tiles: K [key][kc] 2x32KB, V [vc][key] 2x32KB, P 8KB
  __shared__ __align__(16) bf16 Ktile[2][64 * 256];
  __shared__ __align__(16) bf16 Vtile[2][256 * 64];
  __shared__ __align__(16) bf16 Pbuf[4][16 * 64];
  bf16* Pw = &Pbuf[wave][0];

  const bf16* Qn = Qt + (size_t)n * LL * KCH;
  const bf16* Kn = Kt + (size_t)n * LL * KCH;
  const bf16* Vn = Vm + (size_t)n * VCH * LL;

  // Preload Q fragments: A = Q^T (M=16 queries, K=256 channels); served by L2.
  const bf16* qrow = Qn + (size_t)(qbase + l15) * KCH;
  v16bf qa[8];
#pragma unroll
  for (int j = 0; j < 8; ++j) qa[j] = load_frag_a(qrow, j * 32, half);

  v8f oacc[16] = {};
  float mrow[8], lsum[8];
#pragma unroll
  for (int r = 0; r < 8; ++r) { mrow[r] = -3.0e38f; lsum[r] = 0.0f; }
  const float scale = 0.0625f;  // 256^-0.5

  // issue one K+V tile copy (32 b128 async ops per thread)
  auto issue_tile = [&](int kb, int buf) {
    const bf16* gk = Kn + (size_t)kb * KCH;   // 32KB contiguous (Kt rows adjacent)
    uint32_t kl = (uint32_t)(uintptr_t)(&Ktile[buf][0]);
    uint32_t vl = (uint32_t)(uintptr_t)(&Vtile[buf][0]);
#pragma unroll
    for (int i = 0; i < 16; ++i) {
      int c = tid + i * 128;                  // 2048 chunks of 16B
      async_copy_b128(kl + c * 16, gk + (size_t)c * 8);
    }
#pragma unroll
    for (int i = 0; i < 16; ++i) {
      int c = tid + i * 128;                  // 256 rows x 8 chunks
      int row = c >> 3, co = c & 7;
      async_copy_b128(vl + row * 128 + co * 16, Vn + (size_t)row * LL + kb + co * 8);
    }
  };

  issue_tile(0, 0);

  for (int it = 0; it < 64; ++it) {
    int kb = it * 64;
    int cur = it & 1;

    if (it + 1 < 64) {
      issue_tile(kb + 64, cur ^ 1);
      // async loads complete in order: <=32 outstanding => current tile landed
      asm volatile("s_wait_asynccnt 0x20" ::: "memory");
    } else {
      asm volatile("s_wait_asynccnt 0x0" ::: "memory");
    }
    __syncthreads();

    // S tile: 16 queries x 64 keys (4 accumulators), B-frags from LDS
    v8f sacc[4] = {};
#pragma unroll
    for (int t = 0; t < 4; ++t) {
      const bf16* krow = &Ktile[cur][(t * 16 + l15) * 256];
#pragma unroll
      for (int j = 0; j < 8; ++j) {
        v16bf b = load_frag_b(krow, j * 32, half);
        sacc[t] = WMMA_BF16(qa[j], b, sacc[t]);
      }
    }

    // streaming softmax update (rows = queries; 16 lanes of a half share a row)
    float mnew[8], corr[8];
    float p[4][8];
#pragma unroll
    for (int r = 0; r < 8; ++r) {
      float v = fmaxf(fmaxf(sacc[0][r], sacc[1][r]), fmaxf(sacc[2][r], sacc[3][r])) * scale;
#pragma unroll
      for (int m2 = 1; m2 < 16; m2 <<= 1) v = fmaxf(v, __shfl_xor(v, m2, 32));
      mnew[r] = fmaxf(mrow[r], v);
      corr[r] = __expf(mrow[r] - mnew[r]);
    }
#pragma unroll
    for (int t = 0; t < 4; ++t)
#pragma unroll
      for (int r = 0; r < 8; ++r) p[t][r] = __expf(sacc[t][r] * scale - mnew[r]);
#pragma unroll
    for (int r = 0; r < 8; ++r) {
      float s = p[0][r] + p[1][r] + p[2][r] + p[3][r];
#pragma unroll
      for (int m2 = 1; m2 < 16; m2 <<= 1) s += __shfl_xor(s, m2, 32);
      lsum[r] = lsum[r] * corr[r] + s;
      mrow[r] = mnew[r];
    }
#pragma unroll
    for (int tt = 0; tt < 16; ++tt)
#pragma unroll
      for (int r = 0; r < 8; ++r) oacc[tt][r] *= corr[r];

    // P (D-layout) -> LDS -> reload as A fragments (per-wave private tile)
#pragma unroll
    for (int t = 0; t < 4; ++t)
#pragma unroll
      for (int r = 0; r < 8; ++r)
        Pw[(r + half * 8) * 64 + t * 16 + l15] = (bf16)p[t][r];
    asm volatile("s_wait_dscnt 0x0" ::: "memory");

    const bf16* prow = Pw + l15 * 64;
    v16bf pa0 = load_frag_a(prow, 0, half);
    v16bf pa1 = load_frag_a(prow, 32, half);

    // Ctx += P(16q x 64k) * V^T(64k x 256vc), B-frags from LDS
#pragma unroll
    for (int tt = 0; tt < 16; ++tt) {
      const bf16* vrow = &Vtile[cur][(tt * 16 + l15) * 64];
      v16bf b0 = load_frag_b(vrow, 0, half);
      oacc[tt] = WMMA_BF16(pa0, b0, oacc[tt]);
      v16bf b1 = load_frag_b(vrow, 32, half);
      oacc[tt] = WMMA_BF16(pa1, b1, oacc[tt]);
    }

    __syncthreads();  // protect tile buffers before re-issue
  }

  float inv[8];
#pragma unroll
  for (int r = 0; r < 8; ++r) inv[r] = 1.0f / lsum[r];

  bf16* Cn = Ctx + (size_t)n * LL * VCH;
#pragma unroll
  for (int tt = 0; tt < 16; ++tt)
#pragma unroll
    for (int r = 0; r < 8; ++r) {
      int q = qbase + half * 8 + r;
      int vc = tt * 16 + l15;
      Cn[(size_t)q * VCH + vc] = (bf16)(oacc[tt][r] * inv[r]);
    }
}

// ---- output conv: out[512,4096] = wW[512,256] * Ctx^T[256,4096] + bW --------
__global__ __launch_bounds__(256) void outconv_kernel(
    const bf16* __restrict__ Ctx, const bf16* __restrict__ wWb,
    const float* __restrict__ bW, float* __restrict__ out) {
  int lt = blockIdx.x;   // 64 tiles of 64 columns
  int ot = blockIdx.y;   // 4 tiles of 128 rows
  int n  = blockIdx.z;
  int wave = threadIdx.x >> 5;
  int lane = threadIdx.x & 31;
  int half = lane >> 4;
  int l15 = lane & 15;
  int obase = ot * 128 + wave * 16;
  int lbase = lt * 64;

  const bf16* Cn = Ctx + (size_t)n * LL * VCH;
  const bf16* arow = wWb + (size_t)(obase + l15) * VCH;
  const bf16* brow[4];
#pragma unroll
  for (int t = 0; t < 4; ++t) brow[t] = Cn + (size_t)(lbase + t * 16 + l15) * VCH;

  v8f acc[4] = {};
  for (int kc = 0; kc < VCH; kc += 32) {
    v16bf a = load_frag_a(arow, kc, half);
#pragma unroll
    for (int t = 0; t < 4; ++t) {
      v16bf b = load_frag_b(brow[t], kc, half);
      acc[t] = WMMA_BF16(a, b, acc[t]);
    }
  }

  float* On = out + (size_t)n * OCH * LL;
#pragma unroll
  for (int t = 0; t < 4; ++t)
#pragma unroll
    for (int r = 0; r < 8; ++r) {
      int o = obase + half * 8 + r;
      int l = lbase + t * 16 + l15;
      On[(size_t)o * LL + l] = acc[t][r] + bW[o];
    }
}

// ---- host launch ------------------------------------------------------------
extern "C" void kernel_launch(void* const* d_in, const int* in_sizes, int n_in,
                              void* d_out, int out_size, void* d_ws, size_t ws_size,
                              hipStream_t stream) {
  const float* x     = (const float*)d_in[0];
  const float* wq    = (const float*)d_in[1];
  const float* bq    = (const float*)d_in[2];
  const float* gq    = (const float*)d_in[3];
  const float* betaq = (const float*)d_in[4];
  const float* mq    = (const float*)d_in[5];
  const float* vq    = (const float*)d_in[6];
  const float* wk    = (const float*)d_in[7];
  const float* bk    = (const float*)d_in[8];
  const float* gk    = (const float*)d_in[9];
  const float* betak = (const float*)d_in[10];
  const float* mk    = (const float*)d_in[11];
  const float* vk    = (const float*)d_in[12];
  const float* wv    = (const float*)d_in[13];
  const float* bv    = (const float*)d_in[14];
  const float* wW    = (const float*)d_in[15];
  const float* bW    = (const float*)d_in[16];

  char* ws = (char*)d_ws;
  size_t off = 0;
  bf16* Xt = (bf16*)(ws + off);        off += (size_t)NB * LL * CIN * 2;  // 16 MB
  bf16* Wqkv = (bf16*)(ws + off);      off += (size_t)768 * CIN * 2;
  float* biasAll = (float*)(ws + off); off += 1024 * 4;
  bf16* wWb = (bf16*)(ws + off);       off += (size_t)OCH * VCH * 2;
  bf16* Qt = (bf16*)(ws + off);        off += (size_t)NB * LL * KCH * 2;
  bf16* Kt = (bf16*)(ws + off);        off += (size_t)NB * LL * KCH * 2;
  bf16* Vm = (bf16*)(ws + off);        off += (size_t)NB * VCH * LL * 2;
  bf16* Ctx = (bf16*)(ws + off);       off += (size_t)NB * LL * VCH * 2;

  fold_weights<<<1280, 256, 0, stream>>>(wq, bq, gq, betaq, mq, vq,
                                         wk, bk, gk, betak, mk, vk,
                                         wv, bv, wW, Wqkv, biasAll, wWb);
  transpose_x<<<NB * CIN, 256, 0, stream>>>(x, Xt);
  proj_kernel<<<dim3(64, 6, NB), 256, 0, stream>>>(Xt, Wqkv, biasAll, Qt, Kt, Vm);
  attn_kernel<<<dim3(64, NB), 128, 0, stream>>>(Qt, Kt, Vm, Ctx);
  outconv_kernel<<<dim3(64, 4, NB), 256, 0, stream>>>(Ctx, wWb, bW, (float*)d_out);
}
